// DQRN_30734785970915
// MI455X (gfx1250) — matrix operand order repaired and checked by reading, not compiled, not run on hardware
//
#include <hip/hip_runtime.h>
#include <math.h>

typedef __attribute__((ext_vector_type(16))) _Float16 v16h;
typedef __attribute__((ext_vector_type(8)))  _Float16 v8h;
typedef __attribute__((ext_vector_type(4)))  _Float16 v4h;
typedef __attribute__((ext_vector_type(8)))  float    v8f;

static constexpr int NC   = 1024;   // clusters
static constexpr int TT   = 64;     // time steps
static constexpr int DI   = 256;    // input dim
static constexpr int HD   = 128;    // hidden dim (low & high)
static constexpr int G3   = 384;    // 3*HD gate rows
static constexpr int PJ   = 16;     // projection dim
static constexpr int NPAIRS = 523776; // 1024*1023/2

// ---------------------------------------------------------------- helpers
__device__ __forceinline__ float sigm_fast(float x){ return 1.f/(1.f+__expf(-x)); }
__device__ __forceinline__ float tanh_fast(float x){
    x = fminf(fmaxf(x, -15.f), 15.f);
    float e = __expf(2.f*x);
    return (e-1.f)/(e+1.f);
}

__device__ __forceinline__ v8f wmma16(v16h a, v16h b, v8f c){
    // D = A(16x32 f16) * B(32x16 f16) + C(16x16 f32)
    return __builtin_amdgcn_wmma_f32_16x16x32_f16(false, a, false, b, (short)0, c, false, false);
}

// Async global->LDS copy, 16 bytes per lane, tracked by ASYNCcnt (CDNA5 path).
// lds_addr: per-lane LDS byte address (low 32 bits of generic shared pointer)
// goff:     per-lane 32-bit byte offset, gbase: wave-uniform 64-bit base (SGPR pair)
__device__ __forceinline__ void async_ldsb128(unsigned lds_addr, int goff, unsigned long long gbase){
    asm volatile("global_load_async_to_lds_b128 %0, %1, %2"
                 :: "v"(lds_addr), "v"(goff), "s"(gbase) : "memory");
}
__device__ __forceinline__ void wait_async0(){
    asm volatile("s_wait_asynccnt 0x0" ::: "memory");
}

// A fragment (16x32 f16, MxK) from LDS row-major [16][stride] at K-base kb.
// lanes 0-15: row=lane,   elems 0..15 = K kb..kb+15
// lanes 16-31: row=lane-16, elems 0..15 = K kb+16..kb+31
__device__ __forceinline__ v16h load_a_frag_lds(const _Float16* base, int stride, int kb, int lane){
    int row = lane & 15;
    int k0  = kb + ((lane >> 4) << 4);
    const _Float16* p = base + row*stride + k0;
    v8h lo = *(const v8h*)(p);
    v8h hi = *(const v8h*)(p + 8);
    v16h r;
#pragma unroll
    for (int e = 0; e < 8; e++){ r[e] = lo[e]; r[8+e] = hi[e]; }
    return r;
}

// B fragment (32x16 f16, KxN) where B[k][n] = W[n][k], W row-major [N][ldk].
// lane 0-15: col n0+lane, elems = K kb..kb+15; lanes 16-31: K kb+16..kb+31.
__device__ __forceinline__ v16h load_b_frag_g(const _Float16* w, int ldk, int n0, int kb, int lane){
    int n  = n0 + (lane & 15);
    int k0 = kb + ((lane >> 4) << 4);
    const v8h* p = (const v8h*)(w + (size_t)n*ldk + k0);
    v8h lo = p[0], hi = p[1];
    v16h r;
#pragma unroll
    for (int e = 0; e < 8; e++){ r[e] = lo[e]; r[8+e] = hi[e]; }
    return r;
}

// ---------------------------------------------------------------- kernel 0a: convert images to f16
__global__ __launch_bounds__(256) void dqrn_img_cvt(
    const float* __restrict__ in, _Float16* __restrict__ out)
{
    size_t stride = (size_t)gridDim.x * blockDim.x;
    size_t n4 = (size_t)NC*TT*DI/4;
    for (size_t i = (size_t)blockIdx.x*blockDim.x + threadIdx.x; i < n4; i += stride){
        float4 v = ((const float4*)in)[i];
        v4h h; h[0]=(_Float16)v.x; h[1]=(_Float16)v.y; h[2]=(_Float16)v.z; h[3]=(_Float16)v.w;
        ((v4h*)out)[i] = h;
    }
}

// ---------------------------------------------------------------- kernel 0b: prep weights
// Convert GRU-low weights to f16 and pack W_a1[:,16:32]^T into B-fragment order.
__global__ __launch_bounds__(256) void dqrn_prep(
    const float* __restrict__ wih, const float* __restrict__ whh,
    const float* __restrict__ Wa1,
    _Float16* __restrict__ wihf, _Float16* __restrict__ whhf,
    _Float16* __restrict__ bpair)
{
    int stride = gridDim.x * blockDim.x;
    int gid = blockIdx.x * blockDim.x + threadIdx.x;
    for (int i = gid; i < G3*DI; i += stride) wihf[i] = (_Float16)wih[i];
    for (int i = gid; i < G3*HD; i += stride) whhf[i] = (_Float16)whh[i];
    // bpair[tile][lane][e]; tile in {0,1} covers hidden cols 0..15 / 16..31
    for (int i = gid; i < 2*32*16; i += stride){
        int e = i & 15, lane = (i >> 4) & 31, nt = i >> 9;
        int n = nt*16 + (lane & 15);
        int k = (e < 8) ? ((lane >> 4) << 3) + e : 16 + ((lane >> 4) << 3) + (e - 8);          // K position in 32-deep frag
        bpair[i] = (k < 16) ? (_Float16)Wa1[n*32 + 16 + k] : (_Float16)0.f;
    }
}

// ---------------------------------------------------------------- kernel 1: GRU-low
// 64 blocks x 256 threads; each block owns 16 clusters for all 64 steps.
// x tiles are double-buffered and staged with async global->LDS b128 copies.
__global__ __launch_bounds__(256) void dqrn_gru_low(
    const _Float16* __restrict__ imgs16,
    const _Float16* __restrict__ wihf, const _Float16* __restrict__ whhf,
    const float* __restrict__ bih, const float* __restrict__ bhh,
    const float* __restrict__ Wcf, const float* __restrict__ bcf,
    float* __restrict__ rep, float* __restrict__ cproj)
{
    __shared__ __align__(16) _Float16 xs2[2][16*264]; // double-buffered x tiles
    __shared__ __align__(16) _Float16 hs[16*136];     // h tile f16, padded stride
    __shared__ __align__(16) float grz[16*260];       // combined gi+gh for r,z gates
    __shared__ __align__(16) float gni[16*132];       // i_n
    __shared__ __align__(16) float gnh[16*132];       // h_n
    __shared__ float brz[256], bni[128], bnh[128];

    const int tid  = threadIdx.x;
    const int lane = tid & 31;
    const int wave = tid >> 5;
    const int c0   = blockIdx.x * 16;

    brz[tid] = bih[tid] + bhh[tid];
    if (tid < 128){ bni[tid] = bih[256+tid]; bnh[tid] = bhh[256+tid]; }
    for (int i = tid; i < 16*136; i += 256) hs[i] = (_Float16)0.f;

    // Per-thread async-copy slots: 512 x 16B cover one 16x256 f16 tile.
    const unsigned long long gbase =
        (unsigned long long)(uintptr_t)(imgs16 + (size_t)c0 * TT * DI);
    const int m0  = tid >> 5,        cb0 = (tid & 31) << 4;        // slot tid
    const int m1  = (tid + 256)>>5,  cb1 = cb0;                    // slot tid+256
    const unsigned lds0a = (unsigned)(uintptr_t)(&xs2[0][0] + m0*264) + cb0;
    const unsigned lds0b = (unsigned)(uintptr_t)(&xs2[0][0] + m1*264) + cb1;
    const unsigned bufstride = (unsigned)(16*264*sizeof(_Float16));

    // Hoist loop-invariant W_hh B-fragments: 12 x v16h = 96 VGPRs per wave.
    v16h bhR[4], bhZ[4], bhN[4];
#pragma unroll
    for (int kk = 0; kk < 4; kk++){
        bhR[kk] = load_b_frag_g(whhf, HD,       wave*16, kk*32, lane);
        bhZ[kk] = load_b_frag_g(whhf, HD, 128 + wave*16, kk*32, lane);
        bhN[kk] = load_b_frag_g(whhf, HD, 256 + wave*16, kk*32, lane);
    }

    // kick off tile 0
    async_ldsb128(lds0a, (m0*TT + 0)*DI*2 + cb0, gbase);
    async_ldsb128(lds0b, (m1*TT + 0)*DI*2 + cb1, gbase);
    __syncthreads();

    for (int t = 0; t < TT; t++){
        const int cur = t & 1;
        const _Float16* xcur = xs2[cur];
        wait_async0();          // tile t resident in xs2[cur]
        __syncthreads();
        if (t + 1 < TT){        // prefetch tile t+1 into the other buffer
            unsigned dofs = (unsigned)((cur ^ 1) * bufstride);
            async_ldsb128(lds0a + dofs, (m0*TT + t + 1)*DI*2 + cb0, gbase);
            async_ldsb128(lds0b + dofs, (m1*TT + t + 1)*DI*2 + cb1, gbase);
        }

        v8f accR = {}, accZ = {}, accNi = {}, accNh = {};
        // gi = x @ W_ih^T   (K=256, W_ih streamed from L2-resident f16 copy)
#pragma unroll
        for (int kk = 0; kk < 8; kk++){
            v16h a = load_a_frag_lds(xcur, 264, kk*32, lane);
            accR  = wmma16(a, load_b_frag_g(wihf, DI,       wave*16, kk*32, lane), accR);
            accZ  = wmma16(a, load_b_frag_g(wihf, DI, 128 + wave*16, kk*32, lane), accZ);
            accNi = wmma16(a, load_b_frag_g(wihf, DI, 256 + wave*16, kk*32, lane), accNi);
        }
        // gh = h @ W_hh^T   (K=128, B fragments already in registers)
#pragma unroll
        for (int kk = 0; kk < 4; kk++){
            v16h a = load_a_frag_lds(hs, 136, kk*32, lane);
            accR  = wmma16(a, bhR[kk], accR);
            accZ  = wmma16(a, bhZ[kk], accZ);
            accNh = wmma16(a, bhN[kk], accNh);
        }
        // scatter C fragments to LDS (row = v + (lane>>4)*8, col = lane&15)
        {
            int col = lane & 15, rb = (lane >> 4) * 8;
#pragma unroll
            for (int v = 0; v < 8; v++){
                grz[(rb+v)*260 +       wave*16 + col] = accR[v];
                grz[(rb+v)*260 + 128 + wave*16 + col] = accZ[v];
                gni[(rb+v)*132 +       wave*16 + col] = accNi[v];
                gnh[(rb+v)*132 +       wave*16 + col] = accNh[v];
            }
        }
        __syncthreads();
        // gate math: 2048 h-elements, 8 per thread
        {
            int m = tid >> 4, j0 = tid & 15;
#pragma unroll
            for (int u = 0; u < 8; u++){
                int j = j0 + u*16;
                float r = sigm_fast(grz[m*260 + j] + brz[j]);
                float z = sigm_fast(grz[m*260 + 128 + j] + brz[128 + j]);
                float n = tanh_fast(gni[m*132 + j] + bni[j] + r * (gnh[m*132 + j] + bnh[j]));
                float hp = (float)hs[m*136 + j];
                hs[m*136 + j] = (_Float16)((1.f - z)*n + z*hp);
            }
        }
        __syncthreads();
    }

    // emit cluster_rep (f32) and c = relu(h @ W_cf^T + b_cf)
    {
        int m = tid >> 4, j0 = tid & 15;
#pragma unroll
        for (int u = 0; u < 8; u++){
            int j = j0 + u*16;
            rep[(size_t)(c0+m)*HD + j] = (float)hs[m*136 + j];
        }
        int p = tid & 15;
        float acc = bcf[p];
        for (int k = 0; k < HD; k++) acc += (float)hs[m*136 + k] * Wcf[p*HD + k];
        cproj[(c0+m)*PJ + p] = fmaxf(acc, 0.f);
    }
}

// ---------------------------------------------------------------- kernel 2: GRU-high (serial)
__global__ __launch_bounds__(384) void dqrn_gru_high(
    const float* __restrict__ rep,
    const float* __restrict__ wih, const float* __restrict__ whh,
    const float* __restrict__ bih, const float* __restrict__ bhh,
    const float* __restrict__ Wsf, const float* __restrict__ bsf,
    const float* __restrict__ Wa1, const float* __restrict__ ba1,
    float* __restrict__ a0)
{
    __shared__ float x[128], h[128], g[768], ss[16];
    int tid = threadIdx.x;
    if (tid < 128) h[tid] = 0.f;
    const float bi  = bih[tid];
    const float bh_ = bhh[tid];
    const float* wi = wih + tid*HD;     // L2-resident weight rows
    const float* wh = whh + tid*HD;
    __syncthreads();
    for (int i = 0; i < NC; i++){
        if (tid < 128) x[tid] = rep[(size_t)i*HD + tid];
        __syncthreads();
        float a = bi, b = bh_;
#pragma unroll 8
        for (int k = 0; k < HD; k++){ a += wi[k]*x[k]; b += wh[k]*h[k]; }
        g[tid] = a; g[384 + tid] = b;
        __syncthreads();
        if (tid < 128){
            float r = sigm_fast(g[tid] + g[384 + tid]);
            float z = sigm_fast(g[128 + tid] + g[512 + tid]);
            float n = tanh_fast(g[256 + tid] + r * g[640 + tid]);
            h[tid] = (1.f - z)*n + z*h[tid];
        }
        __syncthreads();
    }
    // s = relu(h @ W_sf^T + b_sf); a0 = W_a1[:,0:16] @ s + b_a1 (state part folded)
    if (tid < 16){
        float acc = bsf[tid];
        for (int k = 0; k < HD; k++) acc += Wsf[tid*HD + k] * h[k];
        ss[tid] = fmaxf(acc, 0.f);
    }
    __syncthreads();
    if (tid < 32){
        float acc = ba1[tid];
        for (int k = 0; k < 16; k++) acc += Wa1[tid*32 + k] * ss[k];
        a0[tid] = acc;
    }
}

// ---------------------------------------------------------------- kernel 3: all-pairs q via WMMA
// 4092 blocks x 8 waves; each wave scores 16 pairs with two 16x16x32 WMMAs.
__global__ __launch_bounds__(256) void dqrn_pairs(
    const float* __restrict__ c, const float* __restrict__ a0,
    const _Float16* __restrict__ bpair,
    const float* __restrict__ Wa2, const float* __restrict__ ba2,
    float* __restrict__ q, float* __restrict__ bmax)
{
    __shared__ float red[8][16][16];
    __shared__ float sm[256];
    int tid = threadIdx.x, lane = tid & 31, wave = tid >> 5;
    long p0 = ((long)blockIdx.x * 8 + wave) * 16;
    int row = lane & 15;
    long p = p0 + row;
    // invert p -> (i, j) with p = i*(i-1)/2 + j, j < i  (tril order)
    long i = (long)((1.0 + sqrt(1.0 + 8.0*(double)p)) * 0.5);
    while (i*(i-1)/2 > p) --i;
    while ((i+1)*i/2 <= p) ++i;
    long j = p - i*(i-1)/2;

    // A fragment: merge row (c[i]+c[j]), K=16 zero-padded to 32
    int klo = (lane >> 4) * 8;
    const float* ci = c + i*PJ + klo;
    const float* cj = c + j*PJ + klo;
    v16h a;
#pragma unroll
    for (int e = 0; e < 8; e++){
        a[e]   = (_Float16)(ci[e] + cj[e]);
        a[e+8] = (_Float16)0.f;
    }
    const v8h* pb0 = (const v8h*)(bpair + lane*16);
    const v8h* pb1 = (const v8h*)(bpair + 512 + lane*16);
    v8h l0 = pb0[0], u0 = pb0[1], l1 = pb1[0], u1 = pb1[1];
    v16h b0, b1;
#pragma unroll
    for (int e = 0; e < 8; e++){ b0[e]=l0[e]; b0[8+e]=u0[e]; b1[e]=l1[e]; b1[8+e]=u1[e]; }

    v8f acc0 = {}, acc1 = {};
    acc0 = wmma16(a, b0, acc0);
    acc1 = wmma16(a, b1, acc1);

    // per-lane partial of q = sum_n relu(hid[n]+a0[n]) * W_a2[n]
    int coln = lane & 15, rb = (lane >> 4) * 8;
    float w0 = Wa2[coln],      w1 = Wa2[16 + coln];
    float A0 = a0[coln],       A1 = a0[16 + coln];
#pragma unroll
    for (int v = 0; v < 8; v++){
        red[wave][rb + v][coln] =
            fmaxf(acc0[v] + A0, 0.f) * w0 + fmaxf(acc1[v] + A1, 0.f) * w1;
    }
    __syncthreads();
    float qv = -3.402823466e38f;
    if (lane < 16){
        float s = ba2[0];
        for (int k = 0; k < 16; k++) s += red[wave][lane][k];
        q[p0 + lane] = s;
        qv = s;
    }
    sm[tid] = qv;
    __syncthreads();
    for (int s2 = 128; s2 > 0; s2 >>= 1){
        if (tid < s2) sm[tid] = fmaxf(sm[tid], sm[tid + s2]);
        __syncthreads();
    }
    if (tid == 0) bmax[blockIdx.x] = sm[0];
}

// ---------------------------------------------------------------- softmax tail (deterministic)
__global__ __launch_bounds__(256) void dqrn_reduce_max(
    const float* __restrict__ in, int n, float* __restrict__ out)
{
    __shared__ float sm[256];
    float v = -3.402823466e38f;
    for (int i = threadIdx.x; i < n; i += 256) v = fmaxf(v, in[i]);
    sm[threadIdx.x] = v; __syncthreads();
    for (int s = 128; s > 0; s >>= 1){
        if (threadIdx.x < s) sm[threadIdx.x] = fmaxf(sm[threadIdx.x], sm[threadIdx.x + s]);
        __syncthreads();
    }
    if (threadIdx.x == 0) *out = sm[0];
}

__global__ __launch_bounds__(256) void dqrn_expsum(
    const float* __restrict__ q, const float* __restrict__ qmax,
    float* __restrict__ ebuf, float* __restrict__ bsum)
{
    __shared__ float sm[256];
    int p = blockIdx.x*256 + threadIdx.x;
    float e = 0.f;
    if (p < NPAIRS){ e = __expf(q[p] - *qmax); ebuf[p] = e; }
    sm[threadIdx.x] = e; __syncthreads();
    for (int s = 128; s > 0; s >>= 1){
        if (threadIdx.x < s) sm[threadIdx.x] += sm[threadIdx.x + s];
        __syncthreads();
    }
    if (threadIdx.x == 0) bsum[blockIdx.x] = sm[0];
}

__global__ __launch_bounds__(256) void dqrn_reduce_sum(
    const float* __restrict__ in, int n, float* __restrict__ out)
{
    __shared__ float sm[256];
    float v = 0.f;
    for (int i = threadIdx.x; i < n; i += 256) v += in[i];
    sm[threadIdx.x] = v; __syncthreads();
    for (int s = 128; s > 0; s >>= 1){
        if (threadIdx.x < s) sm[threadIdx.x] += sm[threadIdx.x + s];
        __syncthreads();
    }
    if (threadIdx.x == 0) *out = 1.f / sm[0];
}

__global__ __launch_bounds__(256) void dqrn_norm(
    const float* __restrict__ ebuf, const float* __restrict__ invsum,
    float* __restrict__ out)
{
    int p = blockIdx.x*256 + threadIdx.x;
    if (p < NPAIRS) out[p] = ebuf[p] * (*invsum);
}

// ---------------------------------------------------------------- launch
extern "C" void kernel_launch(void* const* d_in, const int* in_sizes, int n_in,
                              void* d_out, int out_size, void* d_ws, size_t ws_size,
                              hipStream_t stream)
{
    (void)in_sizes; (void)n_in; (void)out_size; (void)ws_size;
    const float* images = (const float*)d_in[0];
    const float* w_ih_l = (const float*)d_in[1];
    const float* w_hh_l = (const float*)d_in[2];
    const float* b_ih_l = (const float*)d_in[3];
    const float* b_hh_l = (const float*)d_in[4];
    const float* w_ih_h = (const float*)d_in[5];
    const float* w_hh_h = (const float*)d_in[6];
    const float* b_ih_h = (const float*)d_in[7];
    const float* b_hh_h = (const float*)d_in[8];
    const float* W_cf   = (const float*)d_in[9];
    const float* b_cf   = (const float*)d_in[10];
    const float* W_sf   = (const float*)d_in[11];
    const float* b_sf   = (const float*)d_in[12];
    const float* W_a1   = (const float*)d_in[13];
    const float* b_a1   = (const float*)d_in[14];
    const float* W_a2   = (const float*)d_in[15];
    const float* b_a2   = (const float*)d_in[16];
    float* out = (float*)d_out;

    // workspace layout (~39 MB total), 256B-aligned slices
    char* ws = (char*)d_ws;
    size_t o = 0;
    auto take = [&](size_t bytes) -> void* {
        void* p = ws + o;
        o = (o + bytes + 255) & ~(size_t)255;
        return p;
    };
    _Float16* imgs16 = (_Float16*)take((size_t)NC*TT*DI*sizeof(_Float16));
    _Float16* wihf   = (_Float16*)take((size_t)G3*DI*sizeof(_Float16));
    _Float16* whhf   = (_Float16*)take((size_t)G3*HD*sizeof(_Float16));
    _Float16* bpair  = (_Float16*)take((size_t)2*32*16*sizeof(_Float16));
    float* rep    = (float*)take((size_t)NC*HD*sizeof(float));
    float* cproj  = (float*)take((size_t)NC*PJ*sizeof(float));
    float* a0     = (float*)take(32*sizeof(float));
    float* q      = (float*)take((size_t)NPAIRS*sizeof(float));
    float* ebuf   = (float*)take((size_t)NPAIRS*sizeof(float));
    float* bmax   = (float*)take(4092*sizeof(float));
    float* bsum   = (float*)take(2046*sizeof(float));
    float* qmax   = (float*)take(sizeof(float));
    float* invsum = (float*)take(sizeof(float));

    dqrn_img_cvt<<<4096, 256, 0, stream>>>(images, imgs16);
    dqrn_prep<<<96, 256, 0, stream>>>(w_ih_l, w_hh_l, W_a1, wihf, whhf, bpair);
    dqrn_gru_low<<<64, 256, 0, stream>>>(imgs16, wihf, whhf, b_ih_l, b_hh_l,
                                         W_cf, b_cf, rep, cproj);
    dqrn_gru_high<<<1, 384, 0, stream>>>(rep, w_ih_h, w_hh_h, b_ih_h, b_hh_h,
                                         W_sf, b_sf, W_a1, b_a1, a0);
    dqrn_pairs<<<4092, 256, 0, stream>>>(cproj, a0, bpair, W_a2, b_a2, q, bmax);
    dqrn_reduce_max<<<1, 256, 0, stream>>>(bmax, 4092, qmax);
    dqrn_expsum<<<2046, 256, 0, stream>>>(q, qmax, ebuf, bsum);
    dqrn_reduce_sum<<<1, 256, 0, stream>>>(bsum, 2046, invsum);
    dqrn_norm<<<2046, 256, 0, stream>>>(ebuf, invsum, out);
}
